// TransformerBlock_16612933501470
// MI455X (gfx1250) — matrix operand order repaired and checked by reading, not compiled, not run on hardware
//
#include <hip/hip_runtime.h>
#include <cstdint>
#include <cstddef>

// ---------------------------------------------------------------------------
// CDNA5 (gfx1250) wave32 WMMA transformer block.
//   x  = x + MSA(LN1(x))          (layer 1)
//   x  = x + MSA(LN2(x))          (layer 2)
//   out= x + FFN(LN3(x))          (relu MLP 1024->4096->1024)
// All GEMMs in bf16 via v_wmma_f32_16x16x32_bf16 (fp32 accumulate).
// Attention is flash-style (online softmax), never materializing T x T.
// GEMM: 64x128 block tile, 32x32 wave tile (4 indep WMMA chains),
//       double-buffered LDS, one barrier per K-step, prefetch hints.
// ---------------------------------------------------------------------------

typedef __attribute__((ext_vector_type(16))) __bf16 v16bf;
typedef __attribute__((ext_vector_type(8)))  float  v8f;

#define DEVINL __device__ __forceinline__

DEVINL unsigned short f2bf(float f) {
  union { float f; unsigned u; } v; v.f = f;
  unsigned r = v.u + 0x7FFFu + ((v.u >> 16) & 1u);   // round-to-nearest-even
  return (unsigned short)(r >> 16);
}

union FragU { v16bf v; unsigned u[8]; };

// A-matrix 16x32 bf16 fragment (ISA 7.12.2): lanes 0-15 hold M=lane;
// VGPR0..3 = K{0,1..6,7} (lanes 0-15) / K{8,9..14,15} (lanes 16-31);
// VGPR4..7 = +16.  Source is row-major [M, K] with leading dim `ld`.
DEVINL v16bf load_a_frag(const unsigned short* base, int ld) {
  const int lane = threadIdx.x & 31;
  const int row  = lane & 15;
  const int koff = (lane >> 4) << 3;           // 0 or 8
  const unsigned short* rp = base + row * ld + koff;
  FragU f;
#pragma unroll
  for (int i = 0; i < 8; ++i) {
    const int kb = ((i & 4) << 2) + 2 * (i & 3);   // {0,2,4,6,16,18,20,22}
    f.u[i] = *(const unsigned*)(rp + kb);
  }
  return f.v;
}

// B-matrix 32x16 bf16 fragment: lane holds column N=lane&15; lanes 0-15 cover
// K=0-7,16-23, lanes 16-31 cover K=8-15,24-31, 2 bf16 per VGPR.  Source is the
// TRANSPOSED operand, row-major [N, K] with leading dim `ld` (contiguous K).
DEVINL v16bf load_b_frag(const unsigned short* base, int ld) {
  const int lane = threadIdx.x & 31;
  const int n    = lane & 15;
  const int koff = (lane >> 4) << 3;           // 0 or 8
  const unsigned short* rp = base + n * ld + koff;
  FragU f;
#pragma unroll
  for (int i = 0; i < 8; ++i) f.u[i] = *(const unsigned*)(rp + ((i & 4) << 2) + 2 * (i & 3));
  return f.v;
}

DEVINL v8f wmma_bf16(v16bf a, v16bf b, v8f c) {
  return __builtin_amdgcn_wmma_f32_16x16x32_bf16(
      /*neg_a=*/false, a, /*neg_b=*/false, b,
      /*c_mod=*/(short)0, c, /*reuse_a=*/false, /*reuse_b=*/false);
}

DEVINL v8f vzero8() { v8f z = {0.f,0.f,0.f,0.f,0.f,0.f,0.f,0.f}; return z; }

template <typename T> DEVINL void store_out(T* p, float v);
template <> DEVINL void store_out<float>(float* p, float v) { *p = v; }
template <> DEVINL void store_out<unsigned short>(unsigned short* p, float v) { *p = f2bf(v); }

// ---------------------------------------------------------------------------
// Weight prep: fp32 [batch][R][Cc] -> bf16 transposed [batch][Cc][R]
// ---------------------------------------------------------------------------
__global__ void convt_k(const float* __restrict__ in, unsigned short* __restrict__ out,
                        int R, int Cc) {
  const size_t base = (size_t)blockIdx.y * R * Cc;
  const int idx = blockIdx.x * 256 + threadIdx.x;
  if (idx >= R * Cc) return;
  const int r = idx / Cc, c = idx - r * Cc;
  out[base + (size_t)c * R + r] = f2bf(in[base + idx]);
}

// ---------------------------------------------------------------------------
// LayerNorm (C=1024) fp32 -> bf16, one row per block (256 thr, 4 elem/thr)
// ---------------------------------------------------------------------------
__global__ __launch_bounds__(256) void ln_k(const float* __restrict__ x,
                                            const float* __restrict__ g,
                                            const float* __restrict__ bt,
                                            unsigned short* __restrict__ o) {
  __shared__ float ssum[8], ssq[8];
  const int row = blockIdx.x;
  const int t = threadIdx.x;
  const float4 v = ((const float4*)(x + (size_t)row * 1024))[t];
  float s  = v.x + v.y + v.z + v.w;
  float s2 = v.x*v.x + v.y*v.y + v.z*v.z + v.w*v.w;
#pragma unroll
  for (int m = 1; m < 32; m <<= 1) { s += __shfl_xor(s, m, 32); s2 += __shfl_xor(s2, m, 32); }
  if ((t & 31) == 0) { ssum[t >> 5] = s; ssq[t >> 5] = s2; }
  __syncthreads();
  float ts = 0.f, ts2 = 0.f;
#pragma unroll
  for (int i = 0; i < 8; ++i) { ts += ssum[i]; ts2 += ssq[i]; }
  const float mean = ts * (1.0f / 1024.0f);
  const float var  = ts2 * (1.0f / 1024.0f) - mean * mean;
  const float rstd = rsqrtf(var + 1e-5f);
  const float4 gg = ((const float4*)g)[t];
  const float4 bb = ((const float4*)bt)[t];
  ushort4 ov;
  ov.x = f2bf((v.x - mean) * rstd * gg.x + bb.x);
  ov.y = f2bf((v.y - mean) * rstd * gg.y + bb.y);
  ov.z = f2bf((v.z - mean) * rstd * gg.z + bb.z);
  ov.w = f2bf((v.w - mean) * rstd * gg.w + bb.w);
  ((ushort4*)(o + (size_t)row * 1024))[t] = ov;
}

// ---------------------------------------------------------------------------
// Tiled WMMA GEMM:  out[M,N] = A[M,K](bf16,row) * Bt[N,K](bf16,row)^T
// Block tile 64x128, K-step 32, 8 waves (2x4), wave tile 32x32
// (2 A-frags x 2 B-frags -> 4 independent WMMA accumulator chains).
// Double-buffered LDS, single barrier per K-step, global prefetch 1 ahead.
// Optional bias / relu / residual(fp32) / transposed store; OutT f32 or bf16.
// ---------------------------------------------------------------------------
template <bool BIAS, bool RELU, bool RES, bool TSTORE, typename OutT>
__global__ __launch_bounds__(256) void gemm_bf16_k(
    const unsigned short* __restrict__ A, int lda,
    const unsigned short* __restrict__ Bt, int ldb,
    OutT* __restrict__ out, int ldo,
    const float* __restrict__ bias,
    const float* __restrict__ res, int ldr,
    int K) {
  __shared__ unsigned short As[2][64 * 40];    // 64 rows x (32 + 8 pad)
  __shared__ unsigned short Bs[2][128 * 40];   // 128 rows x (32 + 8 pad)
  const int tid = threadIdx.x;
  const int m0 = blockIdx.y * 64;
  const int n0 = blockIdx.x * 128;
  const int lrow = tid >> 2;                   // 0..63
  const int lcol = (tid & 3) * 8;              // 0,8,16,24  (8 bf16 = 16B)
  const unsigned short* ag  = A  + (size_t)(m0 + lrow) * lda + lcol;       // 1 chunk
  const unsigned short* bg0 = Bt + (size_t)(n0 + lrow) * ldb + lcol;       // 2 chunks
  const unsigned short* bg1 = Bt + (size_t)(n0 + lrow + 64) * ldb + lcol;
  const int wid = tid >> 5;
  const int wm = wid >> 2;                     // 0..1 -> M offset 32*wm
  const int wn = wid & 3;                      // 0..3 -> N offset 32*wn

  v8f acc00 = vzero8(), acc01 = vzero8(), acc10 = vzero8(), acc11 = vzero8();

  // prologue: fill buffer 0
  {
    const uint4 av  = *(const uint4*)(ag);
    const uint4 bv0 = *(const uint4*)(bg0);
    const uint4 bv1 = *(const uint4*)(bg1);
    *(uint4*)&As[0][lrow * 40 + lcol] = av;
    *(uint4*)&Bs[0][lrow * 40 + lcol] = bv0;
    *(uint4*)&Bs[0][(lrow + 64) * 40 + lcol] = bv1;
  }

  int buf = 0;
  for (int k0 = 0; k0 < K; k0 += 32, buf ^= 1) {
    __syncthreads();   // orders writes(buf) before reads(buf) and
                       // prev reads(buf^1) before this iter's writes(buf^1)
    const bool more = (k0 + 32) < K;
    uint4 av, bv0, bv1;
    if (more) {
      av  = *(const uint4*)(ag  + k0 + 32);    // overlap with WMMA below
      bv0 = *(const uint4*)(bg0 + k0 + 32);
      bv1 = *(const uint4*)(bg1 + k0 + 32);
      if (k0 + 64 < K) {                       // global_prefetch_b8 1 tile ahead
        __builtin_prefetch(ag  + k0 + 64, 0, 1);
        __builtin_prefetch(bg0 + k0 + 64, 0, 1);
        __builtin_prefetch(bg1 + k0 + 64, 0, 1);
      }
    }
    const v16bf a0 = load_a_frag(&As[buf][(wm * 32 +  0) * 40], 40);
    const v16bf a1 = load_a_frag(&As[buf][(wm * 32 + 16) * 40], 40);
    const v16bf b0 = load_b_frag(&Bs[buf][(wn * 32 +  0) * 40], 40);
    const v16bf b1 = load_b_frag(&Bs[buf][(wn * 32 + 16) * 40], 40);
    acc00 = wmma_bf16(a0, b0, acc00);          // 4 independent chains
    acc01 = wmma_bf16(a0, b1, acc01);
    acc10 = wmma_bf16(a1, b0, acc10);
    acc11 = wmma_bf16(a1, b1, acc11);
    if (more) {
      *(uint4*)&As[buf ^ 1][lrow * 40 + lcol] = av;
      *(uint4*)&Bs[buf ^ 1][lrow * 40 + lcol] = bv0;
      *(uint4*)&Bs[buf ^ 1][(lrow + 64) * 40 + lcol] = bv1;
    }
  }

  const int lane = tid & 31;
  const int nl  = lane & 15;
  const int rof = (lane >> 4) * 8;
  v8f acc[2][2] = {{acc00, acc01}, {acc10, acc11}};
#pragma unroll
  for (int ia = 0; ia < 2; ++ia) {
#pragma unroll
    for (int j = 0; j < 2; ++j) {
      const int n = n0 + wn * 32 + j * 16 + nl;
      const float bvv = BIAS ? bias[n] : 0.0f;
#pragma unroll
      for (int r = 0; r < 8; ++r) {
        const int m = m0 + wm * 32 + ia * 16 + rof + r;
        float v = acc[ia][j][r] + bvv;
        if (RELU) v = fmaxf(v, 0.0f);
        if (RES)  v += res[(size_t)m * ldr + n];
        if (TSTORE) store_out(&out[(size_t)n * ldo + m], v);
        else        store_out(&out[(size_t)m * ldo + n], v);
      }
    }
  }
}

// ---------------------------------------------------------------------------
// Flash attention, one (b,h) per block.z/.y, 128 query rows per block,
// 16 rows per wave.  q,k: bf16 [4096,1024] (col = h*64+d).
// vt: bf16 [1024,4096] = V transposed (row = h*64+d, col = b*1024+t).
// out = res + softmax(q k^T / 32) v    (fp32, concat-head layout)
// ---------------------------------------------------------------------------
__global__ __launch_bounds__(256) void attn_k(
    const unsigned short* __restrict__ q,
    const unsigned short* __restrict__ k,
    const unsigned short* __restrict__ vt,
    const float* __restrict__ res,
    float* __restrict__ out) {
  __shared__ unsigned short Pl[8][16 * 32];      // per-wave P staging (D->A relayout)
  const int b = blockIdx.z, h = blockIdx.y;
  const int tid = threadIdx.x;
  const int wid = tid >> 5, lane = tid & 31;
  const int qrow0 = blockIdx.x * 128 + wid * 16;
  const int nl = lane & 15, hi = lane >> 4;

  const unsigned short* qp = q + (size_t)(b * 1024 + qrow0) * 1024 + h * 64;
  const v16bf aq0 = load_a_frag(qp,      1024);  // d 0..31
  const v16bf aq1 = load_a_frag(qp + 32, 1024);  // d 32..63

  float m_i[8], l_i[8];
#pragma unroll
  for (int r = 0; r < 8; ++r) { m_i[r] = -3.0e38f; l_i[r] = 0.0f; }
  v8f o[4];
#pragma unroll
  for (int j = 0; j < 4; ++j) o[j] = vzero8();

  unsigned short* pp = &Pl[wid][0];

  for (int s0 = 0; s0 < 1024; s0 += 32) {
    const unsigned short* kp = k + (size_t)(b * 1024 + s0) * 1024 + h * 64;
    // S[16 x 32] = Q(16x64) K^T : B-fragment rows are key rows (already "Bt")
    v8f sa = vzero8(), sb = vzero8();
    sa = wmma_bf16(aq0, load_b_frag(kp,                  1024), sa);
    sa = wmma_bf16(aq1, load_b_frag(kp + 32,             1024), sa);
    sb = wmma_bf16(aq0, load_b_frag(kp + 16 * 1024,      1024), sb);
    sb = wmma_bf16(aq1, load_b_frag(kp + 16 * 1024 + 32, 1024), sb);

#pragma unroll
    for (int r = 0; r < 8; ++r) {
      const float xa = sa[r] * 0.03125f;           // C^-0.5 = 1/32
      const float xb = sb[r] * 0.03125f;
      float mx = fmaxf(xa, xb);
#pragma unroll
      for (int msk = 1; msk < 16; msk <<= 1) mx = fmaxf(mx, __shfl_xor(mx, msk, 16));
      const float nm = fmaxf(m_i[r], mx);
      const float pa = __expf(xa - nm);
      const float pb = __expf(xb - nm);
      float rs = pa + pb;
#pragma unroll
      for (int msk = 1; msk < 16; msk <<= 1) rs += __shfl_xor(rs, msk, 16);
      const float alpha = __expf(m_i[r] - nm);
      l_i[r] = l_i[r] * alpha + rs;
      m_i[r] = nm;
#pragma unroll
      for (int j = 0; j < 4; ++j) o[j][r] *= alpha;
      const int row = r + 8 * hi;                   // D-layout row for this lane half
      pp[row * 32 +      nl] = f2bf(pa);
      pp[row * 32 + 16 + nl] = f2bf(pb);
    }
    asm volatile("s_wait_dscnt 0x0" ::: "memory");  // P visible before A-frag reload
    const v16bf pA = load_a_frag(pp, 32);           // P as 16x32 A fragment
    const unsigned short* vp = vt + (size_t)(h * 64) * 4096 + b * 1024 + s0;
#pragma unroll
    for (int j = 0; j < 4; ++j)                     // O(16x64) += P V
      o[j] = wmma_bf16(pA, load_b_frag(vp + (size_t)(j * 16) * 4096, 4096), o[j]);
  }

#pragma unroll
  for (int j = 0; j < 4; ++j) {
#pragma unroll
    for (int r = 0; r < 8; ++r) {
      const int m = qrow0 + r + 8 * hi;
      const int col = h * 64 + j * 16 + nl;
      const size_t idx = (size_t)(b * 1024 + m) * 1024 + col;
      out[idx] = res[idx] + o[j][r] / l_i[r];
    }
  }
}

// ---------------------------------------------------------------------------
extern "C" void kernel_launch(void* const* d_in, const int* in_sizes, int n_in,
                              void* d_out, int out_size, void* d_ws, size_t ws_size,
                              hipStream_t stream) {
  (void)in_sizes; (void)n_in; (void)out_size; (void)ws_size;
  const float* x   = (const float*)d_in[0];
  const float* Wq1 = (const float*)d_in[1];
  const float* Wk1 = (const float*)d_in[2];
  const float* Wv1 = (const float*)d_in[3];
  const float* Wq2 = (const float*)d_in[4];
  const float* Wk2 = (const float*)d_in[5];
  const float* Wv2 = (const float*)d_in[6];
  const float* g1  = (const float*)d_in[7];
  const float* b1  = (const float*)d_in[8];
  const float* g2  = (const float*)d_in[9];
  const float* b2  = (const float*)d_in[10];
  const float* g3  = (const float*)d_in[11];
  const float* b3  = (const float*)d_in[12];
  const float* W1  = (const float*)d_in[13];
  const float* bf1 = (const float*)d_in[14];
  const float* W2  = (const float*)d_in[15];
  const float* bf2 = (const float*)d_in[16];
  float* outp = (float*)d_out;

  char* w = (char*)d_ws;
  const size_t MB = (size_t)1 << 20;
  unsigned short* wq1t = (unsigned short*)(w +  0 * MB);  // [1024,1024] bf16 (W^T)
  unsigned short* wk1t = (unsigned short*)(w +  2 * MB);
  unsigned short* wv1t = (unsigned short*)(w +  4 * MB);
  unsigned short* wq2t = (unsigned short*)(w +  6 * MB);
  unsigned short* wk2t = (unsigned short*)(w +  8 * MB);
  unsigned short* wv2t = (unsigned short*)(w + 10 * MB);
  unsigned short* w1t  = (unsigned short*)(w + 12 * MB);  // [4096,1024] bf16
  unsigned short* w2t  = (unsigned short*)(w + 20 * MB);  // [1024,4096] bf16
  unsigned short* xn   = (unsigned short*)(w + 28 * MB);  // LN output bf16 [4096,1024]
  unsigned short* qb   = (unsigned short*)(w + 36 * MB);  // q bf16 [4096,1024]
  unsigned short* kb   = (unsigned short*)(w + 44 * MB);  // k bf16 [4096,1024]
  unsigned short* vtb  = (unsigned short*)(w + 52 * MB);  // v^T bf16 [1024,4096]
  float*          x1   = (float*)(w + 60 * MB);           // after layer-1 residual
  float*          x2   = (float*)(w + 76 * MB);           // after layer-2 residual
  unsigned short* h1   = (unsigned short*)(w + 92 * MB);  // relu hidden bf16 [4096,4096]

  const dim3 blk(256);
  // --- weight prep (bf16, transposed for contiguous B-fragments) ---
  convt_k<<<dim3((1024 * 64 + 255) / 256, 16), blk, 0, stream>>>(Wq1, wq1t, 1024, 64);
  convt_k<<<dim3((1024 * 64 + 255) / 256, 16), blk, 0, stream>>>(Wk1, wk1t, 1024, 64);
  convt_k<<<dim3((1024 * 64 + 255) / 256, 16), blk, 0, stream>>>(Wv1, wv1t, 1024, 64);
  convt_k<<<dim3((1024 * 64 + 255) / 256, 16), blk, 0, stream>>>(Wq2, wq2t, 1024, 64);
  convt_k<<<dim3((1024 * 64 + 255) / 256, 16), blk, 0, stream>>>(Wk2, wk2t, 1024, 64);
  convt_k<<<dim3((1024 * 64 + 255) / 256, 16), blk, 0, stream>>>(Wv2, wv2t, 1024, 64);
  convt_k<<<dim3((1024 * 4096 + 255) / 256, 1), blk, 0, stream>>>(W1, w1t, 1024, 4096);
  convt_k<<<dim3((4096 * 1024 + 255) / 256, 1), blk, 0, stream>>>(W2, w2t, 4096, 1024);

  // --- layer 1: x1 = x + MSA(LN1(x)) ---
  ln_k<<<4096, blk, 0, stream>>>(x, g1, b1, xn);
  gemm_bf16_k<false,false,false,false,unsigned short><<<dim3(8,64), blk, 0, stream>>>(
      xn,1024, wq1t,1024, qb,1024, nullptr, nullptr,0, 1024);
  gemm_bf16_k<false,false,false,false,unsigned short><<<dim3(8,64), blk, 0, stream>>>(
      xn,1024, wk1t,1024, kb,1024, nullptr, nullptr,0, 1024);
  gemm_bf16_k<false,false,false,true ,unsigned short><<<dim3(8,64), blk, 0, stream>>>(
      xn,1024, wv1t,1024, vtb,4096, nullptr, nullptr,0, 1024);
  attn_k<<<dim3(8,16,4), blk, 0, stream>>>(qb, kb, vtb, x, x1);

  // --- layer 2: x2 = x1 + MSA(LN2(x1)) ---
  ln_k<<<4096, blk, 0, stream>>>(x1, g2, b2, xn);
  gemm_bf16_k<false,false,false,false,unsigned short><<<dim3(8,64), blk, 0, stream>>>(
      xn,1024, wq2t,1024, qb,1024, nullptr, nullptr,0, 1024);
  gemm_bf16_k<false,false,false,false,unsigned short><<<dim3(8,64), blk, 0, stream>>>(
      xn,1024, wk2t,1024, kb,1024, nullptr, nullptr,0, 1024);
  gemm_bf16_k<false,false,false,true ,unsigned short><<<dim3(8,64), blk, 0, stream>>>(
      xn,1024, wv2t,1024, vtb,4096, nullptr, nullptr,0, 1024);
  attn_k<<<dim3(8,16,4), blk, 0, stream>>>(qb, kb, vtb, x1, x2);

  // --- FFN: out = x2 + relu(LN3(x2) W1 + bf1) W2 + bf2 ---
  ln_k<<<4096, blk, 0, stream>>>(x2, g3, b3, xn);
  gemm_bf16_k<true ,true ,false,false,unsigned short><<<dim3(32,64), blk, 0, stream>>>(
      xn,1024, w1t,1024, h1,4096, bf1, nullptr,0, 1024);
  gemm_bf16_k<true ,false,true ,false,float><<<dim3(8,64), blk, 0, stream>>>(
      h1,4096, w2t,4096, outp,1024, bf2, x2,1024, 4096);
}
